// MutationGNN_81664508166296
// MI455X (gfx1250) — matrix operand order repaired and checked by reading, not compiled, not run on hardware
//
#include <hip/hip_runtime.h>
#include <hip/hip_bf16.h>
#include <cstdint>

// ---------------- problem constants ----------------
#define N_NODES 20000
#define N_EDGES 80000
#define E_TOT   (N_EDGES + N_NODES)   // + self loops
#define BATCH   64
#define N_MAPS  128
#define HID     128
#define HEADS   8
#define DMODEL  1024                  // HID*HEADS
#define EPSF    1e-5f

typedef __bf16 bf16_t;
typedef bf16_t v16bf __attribute__((ext_vector_type(16)));
typedef float  v8f   __attribute__((ext_vector_type(8)));

// ---------------- helpers ----------------
__device__ __forceinline__ float gelu_f(float x) {
    return 0.5f * x * (1.0f + erff(x * 0.70710678118654752f));
}
// order-preserving float <-> uint encoding for atomicMax on floats
__device__ __forceinline__ unsigned fenc(float f) {
    int i = __float_as_int(f);
    return (i >= 0) ? ((unsigned)i | 0x80000000u) : ~(unsigned)i;
}
__device__ __forceinline__ float fdec(unsigned u) {
    int i = (u & 0x80000000u) ? (int)(u & 0x7fffffffu) : ~(int)u;
    return __int_as_float(i);
}

// ---------------- elementwise / conversion ----------------
__global__ void k_f32_to_bf16(const float* __restrict__ src, bf16_t* __restrict__ dst, int n) {
    int t = blockIdx.x * blockDim.x + threadIdx.x;
    if (t < n) dst[t] = (bf16_t)src[t];
}

__global__ void k_zero_f32(float* p, int n) {
    int t = blockIdx.x * blockDim.x + threadIdx.x;
    if (t < n) p[t] = 0.0f;
}

// gat layer 0: h = x @ W0, with x:[N,1], W0:[1,D]
__global__ void k_gat0_embed(const float* __restrict__ x, const float* __restrict__ W,
                             float* __restrict__ h) {
    int t = blockIdx.x * blockDim.x + threadIdx.x;
    if (t < N_NODES * DMODEL) h[t] = x[t >> 10] * W[t & (DMODEL - 1)];
}

// ---------------- GAT attention ----------------
// per-node, per-head projections: as_[n,h] = sum_k h[n,h,k]*a_src[h,k]
__global__ void k_attn_coef(const float* __restrict__ h, const float* __restrict__ a_src,
                            const float* __restrict__ a_dst, float* __restrict__ as_,
                            float* __restrict__ ad_) {
    int n = blockIdx.x;
    int head = threadIdx.x >> 5, lane = threadIdx.x & 31;
    float s = 0.f, d = 0.f;
    const float* hp = h + (size_t)n * DMODEL + head * HID;
    const float* ap = a_src + head * HID;
    const float* bp = a_dst + head * HID;
    for (int k = lane; k < HID; k += 32) {
        float v = hp[k];
        s += v * ap[k];
        d += v * bp[k];
    }
    for (int off = 16; off > 0; off >>= 1) {
        s += __shfl_down(s, off);
        d += __shfl_down(d, off);
    }
    if (lane == 0) {
        as_[n * HEADS + head] = s;
        ad_[n * HEADS + head] = d;
    }
}

__global__ void k_init_mz(unsigned* __restrict__ m, float* __restrict__ z, int n) {
    int t = blockIdx.x * blockDim.x + threadIdx.x;
    if (t < n) { m[t] = fenc(-INFINITY); z[t] = 0.f; }
}

__device__ __forceinline__ void edge_sd(const int* ei, int e, int& s, int& d) {
    if (e < N_EDGES) { s = ei[e]; d = ei[N_EDGES + e]; }
    else             { s = e - N_EDGES; d = e - N_EDGES; }
}

__global__ void k_edge_max(const int* __restrict__ ei, const float* __restrict__ as_,
                           const float* __restrict__ ad_, float* __restrict__ ebuf,
                           unsigned* __restrict__ m) {
    int t = blockIdx.x * blockDim.x + threadIdx.x;
    if (t >= E_TOT * HEADS) return;
    int e = t >> 3, head = t & 7;
    int s, d; edge_sd(ei, e, s, d);
    float v = as_[s * HEADS + head] + ad_[d * HEADS + head];
    v = (v > 0.f) ? v : 0.2f * v;           // leaky_relu(0.2)
    ebuf[t] = v;
    atomicMax(&m[d * HEADS + head], fenc(v));
}

__global__ void k_edge_exp(const int* __restrict__ ei, const float* __restrict__ ebuf,
                           const unsigned* __restrict__ m, float* __restrict__ pbuf,
                           float* __restrict__ z) {
    int t = blockIdx.x * blockDim.x + threadIdx.x;
    if (t >= E_TOT * HEADS) return;
    int e = t >> 3, head = t & 7;
    int s, d; edge_sd(ei, e, s, d);
    (void)s;
    float p = expf(ebuf[t] - fdec(m[d * HEADS + head]));
    pbuf[t] = p;
    atomicAdd(&z[d * HEADS + head], p);
}

// one block per edge; 256 threads sweep the 1024 feature dims
__global__ void k_edge_aggr(const int* __restrict__ ei, const float* __restrict__ pbuf,
                            const float* __restrict__ z, const float* __restrict__ h,
                            float* __restrict__ agg) {
    int e = blockIdx.x;
    int s, d; edge_sd(ei, e, s, d);
    const float* hs = h + (size_t)s * DMODEL;
    float* od = agg + (size_t)d * DMODEL;
    for (int t = threadIdx.x; t < DMODEL; t += 256) {
        int head = t >> 7;
        float alpha = pbuf[e * HEADS + head] / z[d * HEADS + head];
        atomicAdd(&od[t], alpha * hs[t]);
    }
}

// ---------------- fused bias/GELU/LayerNorm/residual ----------------
#define F_GELU_PRE  1
#define F_LN        2
#define F_GELU_POST 4
__global__ void k_ln_act(const float* __restrict__ in, const float* __restrict__ bias,
                         const float* __restrict__ resid, const float* __restrict__ g,
                         const float* __restrict__ b, float* __restrict__ out,
                         int Dd, int flags) {
    __shared__ float s1[256], s2[256];
    int row = blockIdx.x, tid = threadIdx.x;
    const float* ip = in + (size_t)row * Dd;
    float* op = out + (size_t)row * Dd;
    float mean = 0.f, var = 0.f;
    if (flags & F_LN) {
        float sum = 0.f, sq = 0.f;
        for (int t = tid; t < Dd; t += 256) {
            float v = ip[t] + (bias ? bias[t] : 0.f);
            if (flags & F_GELU_PRE) v = gelu_f(v);
            sum += v; sq += v * v;
        }
        s1[tid] = sum; s2[tid] = sq;
        __syncthreads();
        for (int off = 128; off > 0; off >>= 1) {
            if (tid < off) { s1[tid] += s1[tid + off]; s2[tid] += s2[tid + off]; }
            __syncthreads();
        }
        mean = s1[0] / (float)Dd;
        var  = s2[0] / (float)Dd - mean * mean;
    }
    float rstd = (flags & F_LN) ? rsqrtf(var + EPSF) : 0.f;
    for (int t = tid; t < Dd; t += 256) {
        float v = ip[t] + (bias ? bias[t] : 0.f);
        if (flags & F_GELU_PRE) v = gelu_f(v);
        float y = (flags & F_LN) ? (g[t] * (v - mean) * rstd + b[t]) : v;
        if (flags & F_GELU_POST) y = gelu_f(y);
        op[t] = (resid ? resid[(size_t)row * Dd + t] : 0.f) + y;
    }
}

// ---------------- segment pooling ----------------
__global__ void k_init_pool(float* __restrict__ padd, unsigned* __restrict__ pmax,
                            float* __restrict__ counts) {
    int t = blockIdx.x * blockDim.x + threadIdx.x;
    if (t < BATCH * DMODEL) { padd[t] = 0.f; pmax[t] = fenc(-INFINITY); }
    if (t < BATCH) counts[t] = 0.f;
}

__global__ void k_pool(const float* __restrict__ x, const int* __restrict__ batch,
                       float* __restrict__ padd, unsigned* __restrict__ pmax,
                       float* __restrict__ counts) {
    int t = blockIdx.x * blockDim.x + threadIdx.x;
    if (t >= N_NODES * DMODEL) return;
    int n = t >> 10, d = t & (DMODEL - 1);
    int b = batch[n];
    float v = x[t];
    atomicAdd(&padd[b * DMODEL + d], v);
    atomicMax(&pmax[b * DMODEL + d], fenc(v));
    if (d == 0) atomicAdd(&counts[b], 1.f);
}

// node = concat([mean, max, add], 1)  : [B, 3D]
__global__ void k_build_node(const float* __restrict__ padd, const unsigned* __restrict__ pmax,
                             const float* __restrict__ counts, float* __restrict__ node) {
    int t = blockIdx.x * blockDim.x + threadIdx.x;
    if (t >= BATCH * 3 * DMODEL) return;
    int b = t / (3 * DMODEL), c = t % (3 * DMODEL);
    float v;
    if (c < DMODEL)          v = padd[b * DMODEL + c] / counts[b];
    else if (c < 2 * DMODEL) v = fdec(pmax[b * DMODEL + c - DMODEL]);
    else                     v = padd[b * DMODEL + c - 2 * DMODEL];
    node[t] = v;
}

__global__ void k_gather_map(const float* __restrict__ emb, const int* __restrict__ map_id,
                             float* __restrict__ out) {
    int t = blockIdx.x * blockDim.x + threadIdx.x;
    if (t >= BATCH * 4 * HID) return;
    int b = t >> 9, j = t & (4 * HID - 1);
    out[t] = emb[map_id[b] * (4 * HID) + j];
}

__global__ void k_concat3(const float* __restrict__ a, const float* __restrict__ bm,
                          const float* __restrict__ c, float* __restrict__ out) {
    int t = blockIdx.x * blockDim.x + threadIdx.x;
    if (t >= BATCH * 3 * DMODEL) return;
    int b = t / (3 * DMODEL), col = t % (3 * DMODEL);
    float v;
    if (col < DMODEL)          v = a[b * DMODEL + col];
    else if (col < 2 * DMODEL) v = bm[b * DMODEL + col - DMODEL];
    else                       v = c[b * DMODEL + col - 2 * DMODEL];
    out[t] = v;
}

// final tiny projection [64,256]@[256,10]+b
__global__ void k_fc3(const float* __restrict__ f2, const float* __restrict__ W,
                      const float* __restrict__ bias, float* __restrict__ out) {
    int b = blockIdx.x, j = threadIdx.x;
    if (j >= 10) return;
    float s = bias[j];
    const float* fp = f2 + b * 256;
    for (int k = 0; k < 256; k++) s += fp[k] * W[k * 10 + j];
    out[b * 10 + j] = s;
}

// ---------------- WMMA bf16 GEMM: C[M,N] = A[M,K]@B[K,N] (+bias) ----------------
// 128 threads = 4 waves; block tile 64x64; K-step 32; each wave does a 32x32
// quadrant as 2x2 v_wmma_f32_16x16x32_bf16 tiles.  N,K multiples of 64/32.
#define TM 64
#define TN 64
#define TK 32
__global__ __launch_bounds__(128) void k_gemm_bf16(const bf16_t* __restrict__ A,
                                                   const bf16_t* __restrict__ Bw,
                                                   const float* __restrict__ bias,
                                                   float* __restrict__ C,
                                                   int M, int K, int N) {
    __shared__ bf16_t As[TM][TK + 2];   // row-major [m][k]
    __shared__ bf16_t Bs[TN][TK + 2];   // transposed [n][k] so K is contiguous
    int tid = threadIdx.x;
    int lane = tid & 31, w = tid >> 5;
    int m0 = blockIdx.y * TM, n0 = blockIdx.x * TN;
    int wr = (w >> 1) * 32, wc = (w & 1) * 32;
    int half = lane >> 4, l16 = lane & 15;

    v8f acc[2][2];
    #pragma unroll
    for (int i = 0; i < 2; i++)
        #pragma unroll
        for (int j = 0; j < 2; j++)
            #pragma unroll
            for (int r = 0; r < 8; r++) acc[i][j][r] = 0.f;

    for (int k0 = 0; k0 < K; k0 += TK) {
        // ---- stage A: 64 rows x 32 k; 2 threads/row, 16 bf16 (8 dwords) each
        {
            int m = tid >> 1, ks = (tid & 1) * 16;
            uint32_t* s4 = (uint32_t*)&As[m][ks];
            if (m0 + m < M) {
                const uint32_t* g4 = (const uint32_t*)(A + (size_t)(m0 + m) * K + k0 + ks);
                #pragma unroll
                for (int i = 0; i < 8; i++) s4[i] = g4[i];
            } else {
                #pragma unroll
                for (int i = 0; i < 8; i++) s4[i] = 0u;
            }
        }
        // ---- stage B (transpose): 32 k-rows x 64 n; thread -> (k, 16 n's)
        {
            int kk = tid >> 2, ns = (tid & 3) * 16;
            const bf16_t* gp = Bw + (size_t)(k0 + kk) * N + n0 + ns;
            #pragma unroll
            for (int i = 0; i < 16; i++) Bs[ns + i][kk] = gp[i];
        }
        __syncthreads();

        // ---- build fragments per ISA 7.12.2 layouts
        v16bf a[2], b[2];
        #pragma unroll
        for (int t = 0; t < 2; t++) {
            const bf16_t* ap = &As[wr + t * 16 + l16][0];
            #pragma unroll
            for (int i = 0; i < 4; i++) {
                int kA = half * 8 + 2 * i;          // VGPR0-3: K 0..15 group
                a[t][2 * i]     = ap[kA];
                a[t][2 * i + 1] = ap[kA + 1];
                int kB = 16 + half * 8 + 2 * i;     // VGPR4-7: K 16..31 group
                a[t][8 + 2 * i]     = ap[kB];
                a[t][8 + 2 * i + 1] = ap[kB + 1];
            }
            const bf16_t* bp = &Bs[wc + t * 16 + l16][0];
            #pragma unroll
            for (int i = 0; i < 4; i++) {
                int kk2 = half * 8 + 2 * i;         // VGPR0-3: K 0..15 group
                b[t][2 * i]     = bp[kk2];
                b[t][2 * i + 1] = bp[kk2 + 1];
                int kk3 = 16 + half * 8 + 2 * i;    // VGPR4-7: K 16..31 group
                b[t][8 + 2 * i]     = bp[kk3];
                b[t][8 + 2 * i + 1] = bp[kk3 + 1];
            }
        }
        #pragma unroll
        for (int i = 0; i < 2; i++)
            #pragma unroll
            for (int j = 0; j < 2; j++)
                acc[i][j] = __builtin_amdgcn_wmma_f32_16x16x32_bf16(
                    false, a[i], false, b[j], (short)0, acc[i][j], false, false);
        __syncthreads();
    }

    // ---- store: VGPR r -> row r (lanes 0-15) / 8+r (lanes 16-31), col = lane%16
    #pragma unroll
    for (int i = 0; i < 2; i++) {
        int rowbase = m0 + wr + i * 16 + half * 8;
        #pragma unroll
        for (int j = 0; j < 2; j++) {
            int col = n0 + wc + j * 16 + l16;
            float bv = bias ? bias[col] : 0.f;
            #pragma unroll
            for (int r = 0; r < 8; r++) {
                int row = rowbase + r;
                if (row < M) C[(size_t)row * N + col] = acc[i][j][r] + bv;
            }
        }
    }
}

// ---------------- parameter indices (insertion order of _init_params) ----------------
enum {
    P_GAT0_W = 0, P_GAT0_AS, P_GAT0_AD, P_GAT0_B, P_LN0_G, P_LN0_B,
    P_GAT1_W, P_GAT1_AS, P_GAT1_AD, P_GAT1_B, P_LN1_G, P_LN1_B,
    P_GAT2_W, P_GAT2_AS, P_GAT2_AD, P_GAT2_B, P_LN2_G, P_LN2_B,
    P_SKIP0_W, P_SKIP0_B, P_SKIP1_W, P_SKIP1_B,          // dead in reference
    P_MAP_EMB,
    P_MAPP_W, P_MAPP_B, P_MAPP_G, P_MAPP_BB,
    P_NODEP_W, P_NODEP_B, P_NODEP_G, P_NODEP_BB,
    P_FUS_W, P_FUS_B, P_FUS_G, P_FUS_BB,
    P_SA_WQ, P_SA_BQ, P_SA_WK, P_SA_BK, P_SA_WV, P_SA_BV, P_SA_WO, P_SA_BO,
    P_CA_WQ, P_CA_BQ, P_CA_WK, P_CA_BK, P_CA_WV, P_CA_BV, P_CA_WO, P_CA_BO,
    P_FC1_W, P_FC1_B, P_LNF_G, P_LNF_B, P_FC2_W, P_FC2_B, P_FC3_W, P_FC3_B,
    P_COUNT
};

extern "C" void kernel_launch(void* const* d_in, const int* in_sizes, int n_in,
                              void* d_out, int out_size, void* d_ws, size_t ws_size,
                              hipStream_t stream) {
    (void)in_sizes; (void)n_in; (void)out_size; (void)ws_size;
    const float* x     = (const float*)d_in[0];
    const int*   ei    = (const int*)d_in[1];
    const int*   mapid = (const int*)d_in[2];
    const int*   batch = (const int*)d_in[3];
    const float* P[P_COUNT];
    for (int i = 0; i < P_COUNT; i++) P[i] = (const float*)d_in[4 + i];

    // ---- workspace carve-up (256B aligned bump allocator)
    char* wsp = (char*)d_ws;
    size_t off = 0;
    auto alloc = [&](size_t bytes) -> void* {
        void* p = wsp + off;
        off = (off + bytes + 255) & ~(size_t)255;
        return p;
    };
    const size_t NDf = (size_t)N_NODES * DMODEL * sizeof(float);
    bf16_t* Abf   = (bf16_t*)alloc((size_t)N_NODES * DMODEL * sizeof(bf16_t));
    bf16_t* Wbf   = (bf16_t*)alloc((size_t)3 * DMODEL * DMODEL * sizeof(bf16_t) / 1); // fits 3072x1024
    float*  xa    = (float*)alloc(NDf);
    float*  hb    = (float*)alloc(NDf);
    float*  agg   = (float*)alloc(NDf);
    float*  asb   = (float*)alloc((size_t)N_NODES * HEADS * sizeof(float));
    float*  adb   = (float*)alloc((size_t)N_NODES * HEADS * sizeof(float));
    unsigned* menc= (unsigned*)alloc((size_t)N_NODES * HEADS * sizeof(unsigned));
    float*  zbuf  = (float*)alloc((size_t)N_NODES * HEADS * sizeof(float));
    float*  ebuf  = (float*)alloc((size_t)E_TOT * HEADS * sizeof(float));
    float*  pbuf  = (float*)alloc((size_t)E_TOT * HEADS * sizeof(float));
    float*  padd  = (float*)alloc((size_t)BATCH * DMODEL * sizeof(float));
    unsigned* pmax= (unsigned*)alloc((size_t)BATCH * DMODEL * sizeof(unsigned));
    float*  counts= (float*)alloc(BATCH * sizeof(float));
    float*  nodecat=(float*)alloc((size_t)BATCH * 3 * DMODEL * sizeof(float));
    float*  nodep = (float*)alloc((size_t)BATCH * DMODEL * sizeof(float));
    float*  maper = (float*)alloc((size_t)BATCH * 4 * HID * sizeof(float));
    float*  mape  = (float*)alloc((size_t)BATCH * DMODEL * sizeof(float));
    float*  t1    = (float*)alloc((size_t)BATCH * DMODEL * sizeof(float));
    float*  sab   = (float*)alloc((size_t)BATCH * DMODEL * sizeof(float));
    float*  cab   = (float*)alloc((size_t)BATCH * DMODEL * sizeof(float));
    float*  comb  = (float*)alloc((size_t)BATCH * 3 * DMODEL * sizeof(float));
    float*  fused = (float*)alloc((size_t)BATCH * DMODEL * sizeof(float));
    float*  f1    = (float*)alloc((size_t)BATCH * 4 * HID * sizeof(float));
    float*  f2    = (float*)alloc((size_t)BATCH * 2 * HID * sizeof(float));
    float*  smt   = (float*)alloc((size_t)BATCH * DMODEL * sizeof(float));

    auto cdiv = [](int a, int b) { return (a + b - 1) / b; };

    // bf16-staged WMMA GEMM:  C[M,N] = A@W (+bias)
    auto gemm = [&](const float* Af, const float* Wf, const float* bias, float* Cf,
                    int M, int K, int N) {
        int nA = M * K, nW = K * N;
        k_f32_to_bf16<<<cdiv(nA, 256), 256, 0, stream>>>(Af, Abf, nA);
        k_f32_to_bf16<<<cdiv(nW, 256), 256, 0, stream>>>(Wf, Wbf, nW);
        dim3 grid(N / TN, cdiv(M, TM));
        k_gemm_bf16<<<grid, 128, 0, stream>>>(Abf, Wbf, bias, Cf, M, K, N);
    };

    // one GAT attention+aggregate+finalize pass over h -> xa
    auto gat_attn = [&](const float* h, const float* a_s, const float* a_d,
                        const float* gbias, const float* lg, const float* lb,
                        bool residual) {
        k_attn_coef<<<N_NODES, 256, 0, stream>>>(h, a_s, a_d, asb, adb);
        k_init_mz<<<cdiv(N_NODES * HEADS, 256), 256, 0, stream>>>(menc, zbuf, N_NODES * HEADS);
        k_edge_max<<<cdiv(E_TOT * HEADS, 256), 256, 0, stream>>>(ei, asb, adb, ebuf, menc);
        k_edge_exp<<<cdiv(E_TOT * HEADS, 256), 256, 0, stream>>>(ei, ebuf, menc, pbuf, zbuf);
        k_zero_f32<<<cdiv(N_NODES * DMODEL, 256), 256, 0, stream>>>(agg, N_NODES * DMODEL);
        k_edge_aggr<<<E_TOT, 256, 0, stream>>>(ei, pbuf, zbuf, h, agg);
        // xa = (residual ? xa : 0) + ln(gelu(agg + bias))
        k_ln_act<<<N_NODES, 256, 0, stream>>>(agg, gbias, residual ? xa : nullptr,
                                              lg, lb, xa, DMODEL, F_GELU_PRE | F_LN);
    };

    // ---- GAT layer 0 (K=1 -> scalar embed) ----
    k_gat0_embed<<<cdiv(N_NODES * DMODEL, 256), 256, 0, stream>>>(x, P[P_GAT0_W], hb);
    gat_attn(hb, P[P_GAT0_AS], P[P_GAT0_AD], P[P_GAT0_B], P[P_LN0_G], P[P_LN0_B], false);
    // ---- GAT layer 1 ----
    gemm(xa, P[P_GAT1_W], nullptr, hb, N_NODES, DMODEL, DMODEL);
    gat_attn(hb, P[P_GAT1_AS], P[P_GAT1_AD], P[P_GAT1_B], P[P_LN1_G], P[P_LN1_B], true);
    // ---- GAT layer 2 ----  (skip{i} projections are dead code in the reference)
    gemm(xa, P[P_GAT2_W], nullptr, hb, N_NODES, DMODEL, DMODEL);
    gat_attn(hb, P[P_GAT2_AS], P[P_GAT2_AD], P[P_GAT2_B], P[P_LN2_G], P[P_LN2_B], true);

    // ---- segment pooling -> node features [B, 3D] ----
    k_init_pool<<<cdiv(BATCH * DMODEL, 256), 256, 0, stream>>>(padd, pmax, counts);
    k_pool<<<cdiv(N_NODES * DMODEL, 256), 256, 0, stream>>>(xa, batch, padd, pmax, counts);
    k_build_node<<<cdiv(BATCH * 3 * DMODEL, 256), 256, 0, stream>>>(padd, pmax, counts, nodecat);

    // node_proj: gelu(ln(node @ W + b))
    gemm(nodecat, P[P_NODEP_W], P[P_NODEP_B], smt, BATCH, 3 * DMODEL, DMODEL);
    k_ln_act<<<BATCH, 256, 0, stream>>>(smt, nullptr, nullptr, P[P_NODEP_G], P[P_NODEP_BB],
                                        nodep, DMODEL, F_LN | F_GELU_POST);
    // map_proj: gelu(ln(map_emb[map_id] @ W + b))
    k_gather_map<<<cdiv(BATCH * 4 * HID, 256), 256, 0, stream>>>(P[P_MAP_EMB], mapid, maper);
    gemm(maper, P[P_MAPP_W], P[P_MAPP_B], smt, BATCH, 4 * HID, DMODEL);
    k_ln_act<<<BATCH, 256, 0, stream>>>(smt, nullptr, nullptr, P[P_MAPP_G], P[P_MAPP_BB],
                                        mape, DMODEL, F_LN | F_GELU_POST);

    // seq_len==1 attention: softmax(scores)==1  =>  out = (x@Wv+bv)@Wo+bo
    gemm(nodep, P[P_SA_WV], P[P_SA_BV], t1, BATCH, DMODEL, DMODEL);
    gemm(t1,    P[P_SA_WO], P[P_SA_BO], sab, BATCH, DMODEL, DMODEL);
    gemm(mape,  P[P_CA_WV], P[P_CA_BV], t1, BATCH, DMODEL, DMODEL);
    gemm(t1,    P[P_CA_WO], P[P_CA_BO], cab, BATCH, DMODEL, DMODEL);

    // fusion: gelu(ln([node|sa|ca] @ W + b))
    k_concat3<<<cdiv(BATCH * 3 * DMODEL, 256), 256, 0, stream>>>(nodep, sab, cab, comb);
    gemm(comb, P[P_FUS_W], P[P_FUS_B], smt, BATCH, 3 * DMODEL, DMODEL);
    k_ln_act<<<BATCH, 256, 0, stream>>>(smt, nullptr, nullptr, P[P_FUS_G], P[P_FUS_BB],
                                        fused, DMODEL, F_LN | F_GELU_POST);

    // fc1 -> gelu -> lnf
    gemm(fused, P[P_FC1_W], P[P_FC1_B], smt, BATCH, DMODEL, 4 * HID);
    k_ln_act<<<BATCH, 256, 0, stream>>>(smt, nullptr, nullptr, P[P_LNF_G], P[P_LNF_B],
                                        f1, 4 * HID, F_GELU_PRE | F_LN);
    // fc2 -> gelu
    gemm(f1, P[P_FC2_W], P[P_FC2_B], smt, BATCH, 4 * HID, 2 * HID);
    k_ln_act<<<BATCH, 256, 0, stream>>>(smt, nullptr, nullptr, nullptr, nullptr,
                                        f2, 2 * HID, F_GELU_PRE);
    // fc3 (N=10, plain)
    k_fc3<<<BATCH, 64, 0, stream>>>(f2, P[P_FC3_W], P[P_FC3_B], (float*)d_out);
}
